// MG_model_3238405341627
// MI455X (gfx1250) — compile-verified, matrix operand order checked by reading, not run on hardware
//
#include <hip/hip_runtime.h>

typedef __bf16 bf16;
typedef __attribute__((ext_vector_type(8)))  __bf16 bf16x8;
typedef __attribute__((ext_vector_type(16))) __bf16 bf16x16;
typedef __attribute__((ext_vector_type(8)))  float  f32x8;

// ---------------------------------------------------------------- helpers
__device__ __forceinline__ float sigmoidf(float x){ return 1.f/(1.f+__expf(-x)); }
__device__ __forceinline__ float leakyf(float x){ return x < 0.f ? 0.2f*x : x; }

// Concatenate two 16-byte loads into one 16x32 WMMA fragment (per-lane 16 bf16).
__device__ __forceinline__ bf16x16 ld_frag16(const bf16* __restrict__ p)
{
  bf16x8 lo = *(const bf16x8*)p;
  bf16x8 hi = *(const bf16x8*)(p + 16);
  return __builtin_shufflevector(lo, hi, 0,1,2,3,4,5,6,7,8,9,10,11,12,13,14,15);
}

// ---------------------------------------------------------------- templated WMMA GEMM
// C[M,N](f32) = A[M,K](bf16 row-major, stride LDA) * B[N,K](bf16, stride LDB)^T
// (+bias if BIAS)(+=C if ACC).  Static contract: M%64==0, N%256==0, K%32==0.
// Block = 8 waves as 2(M) x 4(N); wave tile = 32x64 -> 8 WMMAs per K-step,
// fully unconditional inner loop (no guards, no exec manipulation).
template<int M, int N, int K, int LDA, int LDB, int LDC, bool ACC, bool BIAS>
__global__ __launch_bounds__(256)
void gemm_t(const bf16* __restrict__ A, const bf16* __restrict__ B,
            float* __restrict__ C, const float* __restrict__ bias)
{
  const int lane  = threadIdx.x & 31;
  const int wave  = threadIdx.x >> 5;
  const int tileM = blockIdx.y * 64 + (wave >> 2) * 32;
  const int tileN = blockIdx.x * 256 + (wave & 3) * 64;
  const int lmod  = lane & 15;
  const int lhalf = lane >> 4;
  const int koff  = (lane & 16) ? 8 : 0;

  f32x8 acc[2][4];
#pragma unroll
  for (int mi = 0; mi < 2; ++mi)
#pragma unroll
    for (int j = 0; j < 4; ++j)
#pragma unroll
      for (int r = 0; r < 8; ++r) acc[mi][j][r] = 0.f;

  const bf16* ap0 = A + (long)(tileM + lmod) * LDA + koff;
  const bf16* ap1 = ap0 + 16L * LDA;
  const bf16* bp  = B + (long)(tileN + lmod) * LDB + koff;

#pragma unroll 2
  for (int k = 0; k < K; k += 32) {
    __builtin_prefetch((const void*)(ap0 + k + 128), 0, 1);  // speculative; OOB dropped
    const bf16x16 a0 = ld_frag16(ap0 + k);
    const bf16x16 a1 = ld_frag16(ap1 + k);
#pragma unroll
    for (int j = 0; j < 4; ++j) {
      const bf16x16 b = ld_frag16(bp + (long)j * 16 * LDB + k);
      acc[0][j] = __builtin_amdgcn_wmma_f32_16x16x32_bf16(
          false, a0, false, b, (short)0, acc[0][j], false, false);
      acc[1][j] = __builtin_amdgcn_wmma_f32_16x16x32_bf16(
          false, a1, false, b, (short)0, acc[1][j], false, false);
    }
  }

#pragma unroll
  for (int mi = 0; mi < 2; ++mi)
#pragma unroll
    for (int j = 0; j < 4; ++j) {
      const int n = tileN + j * 16 + lmod;
      const float bv = BIAS ? bias[n] : 0.f;
#pragma unroll
      for (int r = 0; r < 8; ++r) {
        const int m = tileM + mi * 16 + r + lhalf * 8;
        float v = acc[mi][j][r] + bv;
        const long idx = (long)m * LDC + n;
        if (ACC) v += C[idx];
        C[idx] = v;
      }
    }
}

template<int M, int N, int K, int LDA, int LDB, int LDC, bool ACC, bool BIAS>
static inline void gemm_launch(const bf16* A, const bf16* B, float* C,
                               const float* bias, hipStream_t s)
{
  static_assert(M % 64 == 0 && N % 256 == 0 && K % 32 == 0, "tile contract");
  dim3 grid(N / 256, M / 64);
  gemm_t<M, N, K, LDA, LDB, LDC, ACC, BIAS><<<grid, 256, 0, s>>>(A, B, C, bias);
}

// ---------------------------------------------------------------- misc small kernels
__global__ void cvt_f32_to_bf16(const float* __restrict__ src, bf16* __restrict__ dst, long n)
{
  long i = (long)blockIdx.x * blockDim.x + threadIdx.x;
  long stride = (long)gridDim.x * blockDim.x;
  for (; i < n; i += stride) dst[i] = (bf16)src[i];
}

// Downcast with zero row-padding: first n_src elements copied, rest zeroed.
__global__ void cvt_pad_bf16(const float* __restrict__ src, bf16* __restrict__ dst,
                             long n_src, long n_tot)
{
  long i = (long)blockIdx.x * blockDim.x + threadIdx.x;
  long stride = (long)gridDim.x * blockDim.x;
  for (; i < n_tot; i += stride) dst[i] = (i < n_src) ? (bf16)src[i] : (bf16)0.f;
}

__global__ void pad_bias_kernel(const float* __restrict__ src, float* __restrict__ dst,
                                int n, int n_pad)
{
  const int i = blockIdx.x * blockDim.x + threadIdx.x;
  if (i < n_pad) dst[i] = (i < n) ? src[i] : 0.f;
}

__global__ void zero_f32(float* __restrict__ p, long n)
{
  long i = (long)blockIdx.x * blockDim.x + threadIdx.x;
  long stride = (long)gridDim.x * blockDim.x;
  for (; i < n; i += stride) p[i] = 0.f;
}

// Embedding gather + per-sim mean aggregation + shifted token stream.
__global__ __launch_bounds__(256)
void embed_kernel(const int* __restrict__ pitch_idx, const int* __restrict__ dur_idx,
                  const float* __restrict__ pitch_table, const float* __restrict__ dur_table,
                  const float* __restrict__ note_sos,
                  bf16* __restrict__ pd_bf, bf16* __restrict__ tok_bf, bf16* __restrict__ agg_bf)
{
  const int s = blockIdx.x;
  const int f = threadIdx.x;
  float vals[8];
  float acc = 0.f;
#pragma unroll
  for (int r = 0; r < 8; ++r) {
    const int n = s * 8 + r;
    const float v = (f < 128) ? pitch_table[pitch_idx[n] * 128 + f]
                              : dur_table[dur_idx[n] * 128 + (f - 128)];
    vals[r] = v; acc += v;
    pd_bf[(long)n * 256 + f] = (bf16)v;
  }
  const float mean = acc * 0.125f;
#pragma unroll
  for (int r = 0; r < 8; ++r) {
    const int n = s * 8 + r;
    agg_bf[(long)n * 256 + f] = (bf16)mean;
    const float tv = (r == 0) ? note_sos[f] : vals[r - 1];
    tok_bf[(long)n * 256 + f] = (bf16)tv;
  }
}

// L2-normalize + leaky + per-sim mean pooling + beat embedding concat.
__global__ __launch_bounds__(256)
void sage_post_kernel(const float* __restrict__ h_pre,
                      const float* __restrict__ beat_table, const int* __restrict__ beat_idx,
                      bf16* __restrict__ sim_full_bf)
{
  __shared__ float red[256];
  const int s = blockIdx.x;
  const int f = threadIdx.x;
  float acc = 0.f;
  for (int r = 0; r < 8; ++r) {
    const long n = (long)s * 8 + r;
    const float v = h_pre[n * 256 + f];
    red[f] = v * v; __syncthreads();
    for (int off = 128; off > 0; off >>= 1) { if (f < off) red[f] += red[f + off]; __syncthreads(); }
    const float ss = red[0]; __syncthreads();
    acc += leakyf(v * rsqrtf(fmaxf(ss, 1e-24f)));
  }
  sim_full_bf[(long)s * 384 + f] = (bf16)(acc * 0.125f);
  if (f < 128)
    sim_full_bf[(long)s * 384 + 256 + f] = (bf16)(beat_table[beat_idx[s] * 128 + f]);
}

// Persistent single-workgroup encoder GRU scan: 4096 sequential steps, H=512.
__global__ __launch_bounds__(1024)
void enc_scan_kernel(const float* __restrict__ gi,      // [4096,1536] (incl bih)
                     const bf16* __restrict__ Whh,      // [1536,512]
                     const float* __restrict__ bhh,     // [1536]
                     float* __restrict__ bar_emb,       // [128,512]
                     bf16* __restrict__ bar_emb_bf)
{
  __shared__ float h[512];
  __shared__ float gh[1536];
  const int t = threadIdx.x;
  if (t < 512) h[t] = 0.f;
  __syncthreads();
  for (int step = 0; step < 4096; ++step) {
    const float* girow = gi + (long)step * 1536;
    for (int o = t; o < 1536; o += 1024) {
      const bf16* w = Whh + (long)o * 512;
      float a = bhh[o];
#pragma unroll 4
      for (int k = 0; k < 512; k += 8) {
        bf16x8 wv = *(const bf16x8*)(w + k);
#pragma unroll
        for (int i = 0; i < 8; ++i) a += (float)wv[i] * h[k + i];
      }
      gh[o] = a;
    }
    __syncthreads();
    if (t < 512) {
      const float r  = sigmoidf(girow[t] + gh[t]);
      const float z  = sigmoidf(girow[512 + t] + gh[512 + t]);
      const float nn = tanhf(girow[1024 + t] + r * gh[1024 + t]);
      const float hn = (1.f - z) * nn + z * h[t];
      h[t] = hn;
      if ((step & 31) == 31) {
        const int b = step >> 5;
        bar_emb[b * 512 + t] = hn;
        bar_emb_bf[b * 512 + t] = (bf16)hn;
      }
    }
    __syncthreads();
  }
}

// 2-head self-attention weights, head-averaged. block=query row, thread=key.
__global__ __launch_bounds__(128)
void attn_kernel(const float* __restrict__ qkv, float* __restrict__ w)
{
  __shared__ float red[128];
  const int q = blockIdx.x, k = threadIdx.x;
  const float* qr = qkv + (long)q * 1536;
  const float* kr = qkv + (long)k * 1536 + 512;
  float out = 0.f;
  for (int hd = 0; hd < 2; ++hd) {
    float s = 0.f;
    for (int d = 0; d < 256; ++d) s += qr[hd * 256 + d] * kr[hd * 256 + d];
    s *= (1.f / 16.f);
    red[k] = s; __syncthreads();
    for (int off = 64; off > 0; off >>= 1) { if (k < off) red[k] = fmaxf(red[k], red[k + off]); __syncthreads(); }
    const float mx = red[0]; __syncthreads();
    const float e = __expf(s - mx);
    red[k] = e; __syncthreads();
    for (int off = 64; off > 0; off >>= 1) { if (k < off) red[k] += red[k + off]; __syncthreads(); }
    const float sum = red[0]; __syncthreads();
    out += 0.5f * e / sum;
  }
  w[q * 128 + k] = out;
}

// Exact top-50% threshold via radix binary search on positive-float bit order.
__global__ __launch_bounds__(1024)
void topk_thresh_kernel(const float* __restrict__ w, float* __restrict__ thr_out)
{
  __shared__ int cnt[1024];
  unsigned lo = 0u, hi = 0x7f800000u;
  const int t = threadIdx.x;
  for (int it = 0; it < 31; ++it) {
    const unsigned mid = lo + ((hi - lo + 1u) >> 1);
    int c = 0;
    for (int i = t; i < 128 * 128; i += 1024)
      c += (__float_as_uint(w[i]) >= mid) ? 1 : 0;
    cnt[t] = c; __syncthreads();
    for (int off = 512; off > 0; off >>= 1) { if (t < off) cnt[t] += cnt[t + off]; __syncthreads(); }
    const int total = cnt[0]; __syncthreads();
    if (total >= 8192) lo = mid; else hi = mid - 1u;
  }
  if (t == 0) thr_out[0] = __uint_as_float(lo);
}

__global__ __launch_bounds__(128)
void gcn_prep_kernel(const float* __restrict__ w, const float* __restrict__ thr_p,
                     float* __restrict__ dinv)
{
  const int c = threadIdx.x;
  const float thr = thr_p[0];
  float d = 1.f;  // self loop weight 1
  for (int r = 0; r < 128; ++r) { const float v = w[r * 128 + c]; if (v >= thr) d += v; }
  dinv[c] = rsqrtf(fmaxf(d, 1e-12f));
}

__global__ __launch_bounds__(512)
void gcn_kernel(const float* __restrict__ w, const float* __restrict__ thr_p,
                const float* __restrict__ dinv, const float* __restrict__ xw,
                const float* __restrict__ gcn_b, float* __restrict__ score_acc)
{
  const int c = blockIdx.x, f = threadIdx.x;
  const float thr = thr_p[0];
  const float dc = dinv[c];
  float acc = dc * dc * xw[c * 512 + f];            // self loop
  for (int r = 0; r < 128; ++r) {
    const float v = w[r * 128 + c];
    if (v >= thr) acc += dinv[r] * v * dc * xw[r * 512 + f];
  }
  acc = leakyf(acc + gcn_b[f]);
  atomicAdd(&score_acc[f], acc * (1.f / 128.f));
}

// mu/std/z reparam + z_hid/z_in matvecs, one workgroup.
__global__ __launch_bounds__(1024)
void latent_kernel(const float* __restrict__ score_acc, const float* __restrict__ eps,
                   const float* __restrict__ mu_W, const float* __restrict__ mu_b,
                   const float* __restrict__ std_W, const float* __restrict__ std_b,
                   const float* __restrict__ zh_W, const float* __restrict__ zh_b,
                   const float* __restrict__ zi_W, const float* __restrict__ zi_b,
                   float* __restrict__ z_hid, float* __restrict__ z_in)
{
  __shared__ float s[512], zz[512], tmp[1024];
  const int t = threadIdx.x;
  if (t < 512) s[t] = leakyf(score_acc[t]);
  __syncthreads();
  {
    const float* Wr = (t < 512) ? (mu_W + (long)t * 512) : (std_W + (long)(t - 512) * 512);
    float a = (t < 512) ? mu_b[t] : std_b[t - 512];
    for (int k = 0; k < 512; ++k) a += Wr[k] * s[k];
    tmp[t] = a;
  }
  __syncthreads();
  if (t < 512) zz[t] = tmp[t] + __expf(tmp[512 + t]) * eps[t];
  __syncthreads();
  {
    float a = zh_b[t];
    const float* Wr = zh_W + (long)t * 512;
    for (int k = 0; k < 512; ++k) a += Wr[k] * zz[k];
    z_hid[t] = a;
  }
  if (t < 256) {
    float a = zi_b[t];
    const float* Wr = zi_W + (long)t * 512;
    for (int k = 0; k < 512; ++k) a += Wr[k] * zz[k];
    z_in[t] = a;
  }
}

__global__ void init_sdec_h_kernel(const float* __restrict__ z_hid,
                                   float* __restrict__ Hs, bf16* __restrict__ Hs_bf)
{
  const int i = blockIdx.x * blockDim.x + threadIdx.x;
  if (i < 128 * 1024) { const float v = z_hid[i & 1023]; Hs[i] = v; Hs_bf[i] = (bf16)v; }
}

__global__ void build_xsim_kernel(const bf16* __restrict__ sim_full_bf,
                                  const float* __restrict__ sim_sos,
                                  const float* __restrict__ z_in,
                                  bf16* __restrict__ x_sim_bf)
{
  long i = (long)blockIdx.x * blockDim.x + threadIdx.x;
  const long n = 4096L * 640;
  const long stride = (long)gridDim.x * blockDim.x;
  for (; i < n; i += stride) {
    const int s = (int)(i / 640);
    const int c = (int)(i % 640);
    float v;
    if (c < 384) v = ((s & 31) == 0) ? sim_sos[c] : (float)sim_full_bf[(long)(s - 1) * 384 + c];
    else         v = z_in[c - 384];
    x_sim_bf[i] = (bf16)v;
  }
}

// sim-decoder GRU gates, 128 bars in parallel at time t.
__global__ void sdec_gate_kernel(const float* __restrict__ gi,     // [4096,3072] incl bih
                                 const float* __restrict__ gh,     // [128,3072] incl bhh
                                 float* __restrict__ H, bf16* __restrict__ H_bf,
                                 bf16* __restrict__ sim_dec_bf, int t)
{
  const int i = blockIdx.x * blockDim.x + threadIdx.x;
  if (i >= 128 * 1024) return;
  const int b = i >> 10, j = i & 1023;
  const long s = (long)b * 32 + t;
  const float* gir = gi + s * 3072;
  const float* ghr = gh + (long)b * 3072;
  const float r  = sigmoidf(gir[j] + ghr[j]);
  const float z  = sigmoidf(gir[1024 + j] + ghr[1024 + j]);
  const float nn = tanhf(gir[2048 + j] + r * ghr[2048 + j]);
  const float hn = (1.f - z) * nn + z * H[i];
  H[i] = hn; H_bf[i] = (bf16)hn;
  sim_dec_bf[s * 1024 + j] = (bf16)hn;
}

// note-decoder GRU gates, 4096 sims in parallel at time t.
__global__ void ndec_gate_kernel(const float* __restrict__ gi_t,   // [4096,1536] token part
                                 const float* __restrict__ s_part, // [4096,1536] incl bih
                                 const float* __restrict__ gh,     // [4096,1536] incl bhh
                                 float* __restrict__ Hn, bf16* __restrict__ Hn_bf,
                                 bf16* __restrict__ note_dec_bf, int t)
{
  const int i = blockIdx.x * blockDim.x + threadIdx.x;
  if (i >= 4096 * 512) return;
  const int s = i >> 9, j = i & 511;
  const float* gir = gi_t + (long)s * 1536;
  const float* spr = s_part + (long)s * 1536;
  const float* ghr = gh + (long)s * 1536;
  const float r  = sigmoidf(gir[j] + spr[j] + ghr[j]);
  const float z  = sigmoidf(gir[512 + j] + spr[512 + j] + ghr[512 + j]);
  const float nn = tanhf(gir[1024 + j] + spr[1024 + j] + r * ghr[1024 + j]);
  const float hn = (1.f - z) * nn + z * Hn[i];
  Hn[i] = hn; Hn_bf[i] = (bf16)hn;
  note_dec_bf[((long)s * 8 + t) * 512 + j] = (bf16)hn;
}

// label-smoothed cross entropy: mean over rows of lse - (1-s)*l_y - s*mean(logits)
__global__ void ce_kernel(const float* __restrict__ logits, int ldc, int ncls,
                          const int* __restrict__ labels, int rows,
                          float smooth, float* __restrict__ accum, int slot)
{
  long i = (long)blockIdx.x * blockDim.x + threadIdx.x;
  const long stride = (long)gridDim.x * blockDim.x;
  float local = 0.f;
  for (long r = i; r < rows; r += stride) {
    const float* row = logits + r * ldc;
    float mx = -1e30f, sum = 0.f, tot = 0.f;
    for (int c = 0; c < ncls; ++c) mx = fmaxf(mx, row[c]);
    for (int c = 0; c < ncls; ++c) { sum += __expf(row[c] - mx); tot += row[c]; }
    const float lse = mx + logf(sum);
    local += lse - (1.f - smooth) * row[labels[r]] - smooth * (tot / ncls);
  }
  atomicAdd(&accum[slot], local / rows);
}

__global__ void finalize_kernel(const float* __restrict__ accum, float* __restrict__ out)
{
  const int i = threadIdx.x;
  if (i < 3) out[i] = accum[i];
}

// ---------------------------------------------------------------- host driver
extern "C" void kernel_launch(void* const* d_in, const int* in_sizes, int n_in,
                              void* d_out, int out_size, void* d_ws, size_t ws_size,
                              hipStream_t stream)
{
  (void)in_sizes; (void)n_in; (void)out_size; (void)ws_size;

  // -------- inputs --------
  const int* pitch_idx = (const int*)d_in[0];
  const int* dur_idx   = (const int*)d_in[1];
  const int* beat_idx  = (const int*)d_in[2];
  const float* eps     = (const float*)d_in[9];
  // params (dict insertion order, leaves 10..49)
  const float* pitch_table = (const float*)d_in[10];
  const float* beat_table  = (const float*)d_in[11];
  const float* dur_table   = (const float*)d_in[12];
  const float* sage_bl   = (const float*)d_in[14];
  const float* enc_bih   = (const float*)d_in[18];
  const float* enc_bhh   = (const float*)d_in[19];
  const float* mha_in_b  = (const float*)d_in[21];
  const float* gcn_b     = (const float*)d_in[23];
  const float* mu_W  = (const float*)d_in[24]; const float* mu_b  = (const float*)d_in[25];
  const float* std_W = (const float*)d_in[26]; const float* std_b = (const float*)d_in[27];
  const float* zh_W  = (const float*)d_in[28]; const float* zh_b  = (const float*)d_in[29];
  const float* zi_W  = (const float*)d_in[30]; const float* zi_b  = (const float*)d_in[31];
  const float* sim_sos  = (const float*)d_in[32];
  const float* note_sos = (const float*)d_in[33];
  const float* sdec_bih = (const float*)d_in[36];
  const float* sdec_bhh = (const float*)d_in[37];
  const float* beat_b   = (const float*)d_in[39];
  const float* ndec_bih = (const float*)d_in[42];
  const float* ndec_bhh = (const float*)d_in[43];
  const float* nh_b     = (const float*)d_in[45];
  const float* pitch_b  = (const float*)d_in[47];
  const float* dur_b    = (const float*)d_in[49];
  float* out = (float*)d_out;

  // -------- workspace arena --------
  char* ws = (char*)d_ws;
  size_t off = 0;
  auto A_ = [&](size_t bytes) -> char* {
    char* p = ws + off;
    off += (bytes + 255) & ~(size_t)255;
    return p;
  };
  bf16*  pd_bf     = (bf16*)A_(32768L * 256 * 2);
  bf16*  tok_bf    = (bf16*)A_(32768L * 256 * 2);
  bf16*  agg_bf    = (bf16*)A_(32768L * 256 * 2);
  float* h_pre     = (float*)A_(32768L * 256 * 4);
  bf16*  simfull_bf= (bf16*)A_(4096L * 384 * 2);
  float* gi_enc    = (float*)A_(4096L * 1536 * 4);
  float* bar_emb   = (float*)A_(128L * 512 * 4);
  bf16*  bar_emb_bf= (bf16*)A_(128L * 512 * 2);
  float* qkv       = (float*)A_(128L * 1536 * 4);
  float* attw      = (float*)A_(128L * 128 * 4);
  float* thr       = (float*)A_(256);
  float* dinv      = (float*)A_(256);
  float* xw        = (float*)A_(128L * 512 * 4);
  float* score_acc = (float*)A_(512 * 4);
  float* z_hid     = (float*)A_(1024 * 4);
  float* z_in      = (float*)A_(256 * 4);
  bf16*  xsim_bf   = (bf16*)A_(4096L * 640 * 2);
  float* gi_sim    = (float*)A_(4096L * 3072 * 4);
  float* Hs        = (float*)A_(128L * 1024 * 4);
  bf16*  Hs_bf     = (bf16*)A_(128L * 1024 * 2);
  float* gh_s      = (float*)A_(128L * 3072 * 4);
  bf16*  simdec_bf = (bf16*)A_(4096L * 1024 * 2);
  float* beat_logits=(float*)A_(4096L * 256 * 4);   // padded to 256 cols
  float* Hn        = (float*)A_(4096L * 512 * 4);
  bf16*  Hn_bf     = (bf16*)A_(4096L * 512 * 2);
  float* s_part    = (float*)A_(4096L * 1536 * 4);
  float* gi_t      = (float*)A_(4096L * 1536 * 4);
  float* ghn       = (float*)A_(4096L * 1536 * 4);
  bf16*  notedec_bf= (bf16*)A_(32768L * 512 * 2);
  float* logits    = (float*)A_(32768L * 256 * 4);  // padded to 256 cols (reused)
  float* accum     = (float*)A_(256);
  float* beat_b_p  = (float*)A_(256 * 4);           // zero-padded biases
  float* pitch_b_p = (float*)A_(256 * 4);
  float* dur_b_p   = (float*)A_(256 * 4);
  bf16* WsageL = (bf16*)A_(256L * 256 * 2);
  bf16* WsageR = (bf16*)A_(256L * 256 * 2);
  bf16* WencIh = (bf16*)A_(1536L * 384 * 2);
  bf16* WencHh = (bf16*)A_(1536L * 512 * 2);
  bf16* Wmha   = (bf16*)A_(1536L * 512 * 2);
  bf16* Wgcn   = (bf16*)A_(512L * 512 * 2);
  bf16* WsdIh  = (bf16*)A_(3072L * 640 * 2);
  bf16* WsdHh  = (bf16*)A_(3072L * 1024 * 2);
  bf16* Wbeat  = (bf16*)A_(256L * 1024 * 2);        // padded 64 -> 256 rows
  bf16* Wnh    = (bf16*)A_(512L * 1024 * 2);
  bf16* WndIh  = (bf16*)A_(1536L * 1280 * 2);
  bf16* WndHh  = (bf16*)A_(1536L * 512 * 2);
  bf16* Wpitch = (bf16*)A_(256L * 512 * 2);         // padded 130 -> 256 rows
  bf16* Wdur   = (bf16*)A_(256L * 512 * 2);         // padded 100 -> 256 rows

  auto cvt = [&](const void* src, bf16* dst, long n) {
    int blocks = (int)((n + 255) / 256); if (blocks > 2048) blocks = 2048;
    cvt_f32_to_bf16<<<blocks, 256, 0, stream>>>((const float*)src, dst, n);
  };
  auto cvt_pad = [&](const void* src, bf16* dst, long n_src, long n_tot) {
    int blocks = (int)((n_tot + 255) / 256); if (blocks > 2048) blocks = 2048;
    cvt_pad_bf16<<<blocks, 256, 0, stream>>>((const float*)src, dst, n_src, n_tot);
  };

  // -------- weight downcasts to bf16 (heads zero-padded to 256 out-rows) --------
  cvt(d_in[13], WsageL, 256L * 256);
  cvt(d_in[15], WsageR, 256L * 256);
  cvt(d_in[16], WencIh, 1536L * 384);
  cvt(d_in[17], WencHh, 1536L * 512);
  cvt(d_in[20], Wmha,   1536L * 512);
  cvt(d_in[22], Wgcn,   512L * 512);
  cvt(d_in[34], WsdIh,  3072L * 640);
  cvt(d_in[35], WsdHh,  3072L * 1024);
  cvt(d_in[44], Wnh,    512L * 1024);
  cvt(d_in[40], WndIh,  1536L * 1280);
  cvt(d_in[41], WndHh,  1536L * 512);
  cvt_pad(d_in[38], Wbeat,  64L * 1024,  256L * 1024);
  cvt_pad(d_in[46], Wpitch, 130L * 512,  256L * 512);
  cvt_pad(d_in[48], Wdur,   100L * 512,  256L * 512);
  pad_bias_kernel<<<1, 256, 0, stream>>>(beat_b,  beat_b_p,  64,  256);
  pad_bias_kernel<<<1, 256, 0, stream>>>(pitch_b, pitch_b_p, 130, 256);
  pad_bias_kernel<<<1, 256, 0, stream>>>(dur_b,   dur_b_p,   100, 256);

  zero_f32<<<1, 512, 0, stream>>>(score_acc, 512);
  zero_f32<<<1, 32, 0, stream>>>(accum, 8);

  // -------- embeddings + SAGE --------
  embed_kernel<<<4096, 256, 0, stream>>>(pitch_idx, dur_idx, pitch_table, dur_table,
                                         note_sos, pd_bf, tok_bf, agg_bf);
  gemm_launch<32768, 256, 256, 256, 256, 256, false, true >(agg_bf, WsageL, h_pre, sage_bl, stream);
  gemm_launch<32768, 256, 256, 256, 256, 256, true,  false>(pd_bf,  WsageR, h_pre, nullptr, stream);
  sage_post_kernel<<<4096, 256, 0, stream>>>(h_pre, beat_table, beat_idx, simfull_bf);

  // -------- encoder GRU (input precompute + persistent scan) --------
  gemm_launch<4096, 1536, 384, 384, 384, 1536, false, true>(simfull_bf, WencIh, gi_enc, enc_bih, stream);
  enc_scan_kernel<<<1, 1024, 0, stream>>>(gi_enc, WencHh, enc_bhh, bar_emb, bar_emb_bf);

  // -------- attention + GCN + latent --------
  gemm_launch<128, 1536, 512, 512, 512, 1536, false, true>(bar_emb_bf, Wmha, qkv, mha_in_b, stream);
  attn_kernel<<<128, 128, 0, stream>>>(qkv, attw);
  topk_thresh_kernel<<<1, 1024, 0, stream>>>(attw, thr);
  gcn_prep_kernel<<<1, 128, 0, stream>>>(attw, thr, dinv);
  gemm_launch<128, 512, 512, 512, 512, 512, false, false>(bar_emb_bf, Wgcn, xw, nullptr, stream);
  gcn_kernel<<<128, 512, 0, stream>>>(attw, thr, dinv, xw, gcn_b, score_acc);
  latent_kernel<<<1, 1024, 0, stream>>>(score_acc, eps, mu_W, mu_b, std_W, std_b,
                                        zh_W, zh_b, zi_W, zi_b, z_hid, z_in);

  // -------- sim decoder: 128 bars parallel, 32 sequential steps --------
  init_sdec_h_kernel<<<512, 256, 0, stream>>>(z_hid, Hs, Hs_bf);
  build_xsim_kernel<<<2048, 256, 0, stream>>>(simfull_bf, sim_sos, z_in, xsim_bf);
  gemm_launch<4096, 3072, 640, 640, 640, 3072, false, true>(xsim_bf, WsdIh, gi_sim, sdec_bih, stream);
  for (int t = 0; t < 32; ++t) {
    gemm_launch<128, 3072, 1024, 1024, 1024, 3072, false, true>(Hs_bf, WsdHh, gh_s, sdec_bhh, stream);
    sdec_gate_kernel<<<512, 256, 0, stream>>>(gi_sim, gh_s, Hs, Hs_bf, simdec_bf, t);
  }

  // -------- beat head + loss (padded to 256 classes, CE reads 64) --------
  gemm_launch<4096, 256, 1024, 1024, 1024, 256, false, true>(simdec_bf, Wbeat, beat_logits, beat_b_p, stream);
  ce_kernel<<<64, 256, 0, stream>>>(beat_logits, 256, 64, beat_idx, 4096, 0.0f, accum, 0);

  // -------- note decoder: 4096 sims parallel, 8 sequential steps --------
  gemm_launch<4096, 512, 1024, 1024, 1024, 512, false, true>(simdec_bf, Wnh, Hn, nh_b, stream);
  cvt_f32_to_bf16<<<2048, 256, 0, stream>>>(Hn, Hn_bf, 4096L * 512);
  // sim_dec contribution to gi (cols 256..1279 of ndec_Wih), shared by all 8 notes
  gemm_launch<4096, 1536, 1024, 1024, 1280, 1536, false, true>(simdec_bf, WndIh + 256, s_part, ndec_bih, stream);
  for (int t = 0; t < 8; ++t) {
    // token contribution: rows n = 8*s + t of tok_bf => base offset t*256, lda 8*256
    gemm_launch<4096, 1536, 256, 2048, 1280, 1536, false, false>(tok_bf + (long)t * 256, WndIh, gi_t, nullptr, stream);
    gemm_launch<4096, 1536, 512, 512, 512, 1536, false, true>(Hn_bf, WndHh, ghn, ndec_bhh, stream);
    ndec_gate_kernel<<<8192, 256, 0, stream>>>(gi_t, s_part, ghn, Hn, Hn_bf, notedec_bf, t);
  }

  // -------- pitch / dur heads + losses (padded logits buffer reused) --------
  gemm_launch<32768, 256, 512, 512, 512, 256, false, true>(notedec_bf, Wpitch, logits, pitch_b_p, stream);
  ce_kernel<<<256, 256, 0, stream>>>(logits, 256, 130, pitch_idx, 32768, 0.1f, accum, 1);
  gemm_launch<32768, 256, 512, 512, 512, 256, false, true>(notedec_bf, Wdur, logits, dur_b_p, stream);
  ce_kernel<<<256, 256, 0, stream>>>(logits, 256, 100, dur_idx, 32768, 0.0f, accum, 2);

  finalize_kernel<<<1, 32, 0, stream>>>(accum, out);
}
